// RanDLANet_10273561772203
// MI455X (gfx1250) — matrix operand order repaired and checked by reading, not compiled, not run on hardware
//
#include <hip/hip_runtime.h>
#include <hip/hip_bf16.h>

typedef __attribute__((ext_vector_type(16))) _Float16 v16h;
typedef __attribute__((ext_vector_type(8)))  float    v8f;

#define BATCH  2
#define NPTS   8192
#define KNBR   16
#define CIN    32
#define COUT   64
#define HC     32
#define BN_EPS 1e-6f

// ---- workspace layout (offsets in floats) ----
#define IDX_OFF    ((size_t)0)                      // int[262144]
#define DIST_OFF   ((size_t)262144)                 // f32[262144]
#define XH_OFF     ((size_t)524288)                 // f32[16384*32]
#define Y_OFF      ((size_t)1048576)                // f32[262144*32] (reused stage1/stage2)
#define Q1_OFF     ((size_t)9437184)                // f32[16384*32]
#define Q2_OFF     ((size_t)9961472)                // f32[16384*64]
#define SS_OFF     ((size_t)11010048)               // f32[16384*128]
#define STATS_OFF  ((size_t)13107200)               // f32[576] raw sum/sumsq
#define SCSH_OFF   ((size_t)13107776)               // f32[576] scale/shift
#define WH_OFF     ((size_t)13108352)               // _Float16[2*4096] (B-layout, 2048+2048 floats)

__device__ __forceinline__ float warp_sum32(float v) {
#pragma unroll
  for (int o = 16; o > 0; o >>= 1) v += __shfl_xor(v, o, 32);
  return v;
}

// ---------------------------------------------------------------------------
// prep: convert attention weights to f16 *in WMMA B-register layout*, zero stats
//   B element for (j,t) tile: lane L, elem e  <-  W[32t + (L>>4)*16 + e][16j + (L&15)]
//   stored linearly as wB[((j*2+t)*32 + L)*16 + e]
// ---------------------------------------------------------------------------
__global__ void prep_kernel(const float* __restrict__ w1, const float* __restrict__ w2,
                            _Float16* __restrict__ h1, _Float16* __restrict__ h2,
                            float* __restrict__ stats) {
  const int i = blockIdx.x * 256 + threadIdx.x;
  if (i < 8192) {
    const int o = i & 4095;
    const int e = o & 15;
    const int L = (o >> 4) & 31;
    const int t = (o >> 9) & 1;
    const int j = (o >> 10) & 3;
    const int kk = 32 * t + ((L >> 4) << 4) + e;
    const int nn = 16 * j + (L & 15);
    if (i < 4096) h1[o] = (_Float16)w1[kk * 64 + nn];
    else          h2[o] = (_Float16)w2[kk * 64 + nn];
  }
  if (i < 576) stats[i] = 0.f;
}

// ---------------------------------------------------------------------------
// brute-force KNN (K=16), LDS-tiled candidates, register insertion network
// ---------------------------------------------------------------------------
#define KTILE 512
__global__ void knn_kernel(const float* __restrict__ coords,
                           int* __restrict__ oidx, float* __restrict__ odist) {
  __shared__ float cx[KTILE], cy[KTILE], cz[KTILE], cs[KTILE];
  const int tid = threadIdx.x;
  const int b = blockIdx.x / (NPTS / 256);
  const int n = (blockIdx.x % (NPTS / 256)) * 256 + tid;
  const float* q = coords + ((size_t)b * NPTS + n) * 3;
  const float qx = q[0], qy = q[1], qz = q[2];
  const float qs = qx * qx + qy * qy + qz * qz;
  float bd[16]; int bi[16];
#pragma unroll
  for (int i = 0; i < 16; ++i) { bd[i] = 3.4e38f; bi[i] = 0; }
  for (int m0 = 0; m0 < NPTS; m0 += KTILE) {
    __syncthreads();
    for (int j = tid; j < KTILE; j += 256) {
      const float* t = coords + ((size_t)b * NPTS + m0 + j) * 3;
      const float x = t[0], y = t[1], z = t[2];
      cx[j] = x; cy[j] = y; cz[j] = z; cs[j] = x * x + y * y + z * z;
    }
    __syncthreads();
    for (int j = 0; j < KTILE; ++j) {
      const float dot = qx * cx[j] + qy * cy[j] + qz * cz[j];
      const float d2 = qs + cs[j] - 2.f * dot;
      if (d2 < bd[15]) {           // ascending list; stable ties (strict <)
        const int mi = m0 + j;
#pragma unroll
        for (int p = 15; p > 0; --p) {
          const bool cp = d2 < bd[p];
          const bool cm = d2 < bd[p - 1];
          bd[p] = cp ? (cm ? bd[p - 1] : d2) : bd[p];
          bi[p] = cp ? (cm ? bi[p - 1] : mi) : bi[p];
        }
        if (d2 < bd[0]) { bd[0] = d2; bi[0] = mi; }
      }
    }
  }
  int*   io   = oidx  + ((size_t)b * NPTS + n) * KNBR;
  float* dout = odist + ((size_t)b * NPTS + n) * KNBR;
#pragma unroll
  for (int i = 0; i < 16; ++i) {
    io[i] = bi[i];
    dout[i] = sqrtf(fmaxf(bd[i], 1e-12f));
  }
}

// ---------------------------------------------------------------------------
// mlp1 (leaky 0.2) + shortcut pre-BN (32->128) + shortcut BN stats
// ---------------------------------------------------------------------------
__global__ void point_mlp_kernel(const float* __restrict__ feat,
                                 const float* __restrict__ W1, const float* __restrict__ b1,
                                 const float* __restrict__ Ws, const float* __restrict__ bs,
                                 float* __restrict__ xh, float* __restrict__ sshort,
                                 float* __restrict__ gsum, float* __restrict__ gsq) {
  __shared__ float sW1[1024], sb1[32];
  __shared__ float sWs[4096], sbs[128];
  __shared__ float lsum[128], lsq[128];
  const int tid = threadIdx.x;
  for (int i = tid; i < 1024; i += 256) sW1[i] = W1[i];
  for (int i = tid; i < 4096; i += 256) sWs[i] = Ws[i];
  if (tid < 32) sb1[tid] = b1[tid];
  if (tid < 128) { sbs[tid] = bs[tid]; lsum[tid] = 0.f; lsq[tid] = 0.f; }
  __syncthreads();
  const int pt = blockIdx.x * 256 + tid;
  const int lane = tid & 31;
  float f[32];
#pragma unroll
  for (int c = 0; c < 32; ++c) f[c] = feat[(size_t)pt * 32 + c];
  for (int o = 0; o < 32; ++o) {
    float acc = sb1[o];
#pragma unroll
    for (int c = 0; c < 32; ++c) acc = fmaf(f[c], sW1[c * 32 + o], acc);
    acc = (acc >= 0.f) ? acc : 0.2f * acc;
    xh[(size_t)pt * 32 + o] = acc;
  }
  for (int o = 0; o < 128; ++o) {
    float acc = sbs[o];
#pragma unroll
    for (int c = 0; c < 32; ++c) acc = fmaf(f[c], sWs[c * 128 + o], acc);
    sshort[(size_t)pt * 128 + o] = acc;
    const float s = warp_sum32(acc);
    const float q = warp_sum32(acc * acc);
    if (lane == 0) { atomicAdd(&lsum[o], s); atomicAdd(&lsq[o], q); }
  }
  __syncthreads();
  if (tid < 128) { atomicAdd(&gsum[tid], lsum[tid]); atomicAdd(&gsq[tid], lsq[tid]); }
}

// ---------------------------------------------------------------------------
// LocSE: spatial encoding (10) -> h=32 pre-BN, + BN stats   (one thread / bnk)
// ---------------------------------------------------------------------------
__global__ void locse_kernel(const float* __restrict__ coords,
                             const int* __restrict__ nidx, const float* __restrict__ dist,
                             const float* __restrict__ W, const float* __restrict__ bias,
                             float* __restrict__ y,
                             float* __restrict__ gsum, float* __restrict__ gsq) {
  __shared__ float sW[320], sb[32], lsum[32], lsq[32];
  const int tid = threadIdx.x;
  if (tid < 320) sW[tid] = W[tid];
  if (tid < 32) { sb[tid] = bias[tid]; lsum[tid] = 0.f; lsq[tid] = 0.f; }
  __syncthreads();
  const int g = blockIdx.x * 256 + tid;
  const int b = g / (NPTS * KNBR);
  const int r = g % (NPTS * KNBR);
  const int n = r >> 4;
  const float* cc = coords + ((size_t)b * NPTS + n) * 3;
  const int ni = nidx[g];
  const float* nc = coords + ((size_t)b * NPTS + ni) * 3;
  float se[10];
  se[0] = cc[0]; se[1] = cc[1]; se[2] = cc[2];
  se[3] = nc[0]; se[4] = nc[1]; se[5] = nc[2];
  se[6] = se[0] - se[3]; se[7] = se[1] - se[4]; se[8] = se[2] - se[5];
  se[9] = dist[g];
  float yv[32];
#pragma unroll
  for (int c = 0; c < 32; ++c) yv[c] = sb[c];
#pragma unroll
  for (int j = 0; j < 10; ++j)
#pragma unroll
    for (int c = 0; c < 32; ++c) yv[c] = fmaf(se[j], sW[j * 32 + c], yv[c]);
  float* yo = y + (size_t)g * 32;
  const int lane = tid & 31;
#pragma unroll
  for (int c = 0; c < 32; ++c) {
    yo[c] = yv[c];
    const float s = warp_sum32(yv[c]);
    const float q = warp_sum32(yv[c] * yv[c]);
    if (lane == 0) { atomicAdd(&lsum[c], s); atomicAdd(&lsq[c], q); }
  }
  __syncthreads();
  if (tid < 32) { atomicAdd(&gsum[tid], lsum[tid]); atomicAdd(&gsq[tid], lsq[tid]); }
}

// ---------------------------------------------------------------------------
// BN finalize: scale = g*rsqrt(var+eps); shift = beta - mean*scale
// ---------------------------------------------------------------------------
__global__ void bn_fin_kernel(const float* __restrict__ sum, const float* __restrict__ sq,
                              const float* __restrict__ g, const float* __restrict__ be,
                              float* __restrict__ scale, float* __restrict__ shift,
                              float inv_cnt, int nch) {
  const int c = threadIdx.x;
  if (c < nch) {
    const float m = sum[c] * inv_cnt;
    const float v = sq[c] * inv_cnt - m * m;
    const float sc = g[c] * rsqrtf(v + BN_EPS);
    scale[c] = sc;
    shift[c] = be[c] - m * sc;
  }
}

// ---------------------------------------------------------------------------
// attention pooling: one wave per point.
//   x1(16x64) built in LDS (f32 copy + f16 copy pre-packed in WMMA A layout);
//   logits = x1 @ W_att_lin via v_wmma_f32_16x16x32_f16 with contiguous
//   32B LDS operand loads; softmax over 16 neighbors (shfl_xor 16);
//   p = sum_k softmax*x1 ; q = p @ W_att_mlp + b  (+ BN stats on q)
// ---------------------------------------------------------------------------
__global__ void att_pool_kernel(const float* __restrict__ y,
                                const float* __restrict__ lscale, const float* __restrict__ lshift,
                                const float* __restrict__ xsrc,
                                const float* __restrict__ xscale, const float* __restrict__ xshift,
                                const _Float16* __restrict__ WhB,
                                const float* __restrict__ Wmlp, const float* __restrict__ bmlp,
                                float* __restrict__ qout,
                                float* __restrict__ gsum, float* __restrict__ gsq,
                                int n_out) {
  __shared__ __align__(32) _Float16 sWh[4096];       // B-layout f16 weights
  __shared__ __align__(32) _Float16 sa16[8][1024];   // per-wave A-layout x1 (2 K-steps)
  __shared__ float sx1[8][16 * 64];                  // per-wave x1 f32
  __shared__ float spool[8][64];
  __shared__ float lsum[64], lsq[64];

  const int tid = threadIdx.x;
  const int lane = tid & 31;
  const int wave = tid >> 5;
  const int pt = blockIdx.x * 8 + wave;     // 0..16383

  {
    const float4* src = (const float4*)WhB;          // 4096 halfs = 512 float4... copy as u32x4
    float4* dst = (float4*)sWh;
    for (int i = tid; i < 512; i += 256) dst[i] = src[i];
  }
  if (tid < 64) { lsum[tid] = 0.f; lsq[tid] = 0.f; }

  float* x1t = sx1[wave];
  _Float16* a16 = sa16[wave];
  for (int t = lane; t < 1024; t += 32) {
    const int k = t >> 6;
    const int c = t & 63;
    float v;
    if (c < 32) {
      const float yv = y[((size_t)pt * KNBR + k) * 32 + c];
      v = fmaxf(fmaf(yv, lscale[c], lshift[c]), 0.f);           // relu(bn(se@W))
    } else {
      float raw = xsrc[(size_t)pt * 32 + (c - 32)];
      if (xscale) raw = fmaxf(fmaf(raw, xscale[c - 32], xshift[c - 32]), 0.f);
      v = raw;                                                  // broadcast feature half
    }
    x1t[t] = v;
    // scatter f16 copy into A-register layout: step = c>>5, cc = c&31
    const int cc = c & 31;
    const int hi2 = (cc >> 4) & 1;
    const int e = cc & 15;
    a16[((c >> 5) << 9) + ((k + (hi2 << 4)) << 4) + e] = (_Float16)v;
  }
  __syncthreads();

  const int m  = lane & 15;
  const int hi = (lane >> 4) & 1;
  const v16h a0 = *(const v16h*)(a16 + lane * 16);        // K-step 0, contiguous 32B
  const v16h a1 = *(const v16h*)(a16 + 512 + lane * 16);  // K-step 1

#pragma unroll
  for (int j = 0; j < 4; ++j) {                 // four 16-wide N tiles
    const v16h b0 = *(const v16h*)(sWh + (j * 2 + 0) * 512 + lane * 16);
    const v16h b1 = *(const v16h*)(sWh + (j * 2 + 1) * 512 + lane * 16);
    v8f acc = {};
    acc = __builtin_amdgcn_wmma_f32_16x16x32_f16(false, a0, false, b0, (short)0, acc, false, false);
    acc = __builtin_amdgcn_wmma_f32_16x16x32_f16(false, a1, false, b1, (short)0, acc, false, false);

    // softmax over neighbors (M) within this lane's column j*16+m
    float mx = acc[0];
#pragma unroll
    for (int r = 1; r < 8; ++r) mx = fmaxf(mx, acc[r]);
    mx = fmaxf(mx, __shfl_xor(mx, 16, 32));
    float ex[8]; float s = 0.f;
#pragma unroll
    for (int r = 0; r < 8; ++r) { ex[r] = __expf(acc[r] - mx); s += ex[r]; }
    s += __shfl_xor(s, 16, 32);
    const float inv = 1.f / s;
    float psum = 0.f;
#pragma unroll
    for (int r = 0; r < 8; ++r)
      psum = fmaf(ex[r] * inv, x1t[(hi * 8 + r) * 64 + j * 16 + m], psum);
    psum += __shfl_xor(psum, 16, 32);
    spool[wave][j * 16 + m] = psum;             // both lane halves write same value
  }
  __syncthreads();

  const float* pv = spool[wave];
  for (int o = lane; o < n_out; o += 32) {
    float qacc = bmlp[o];
#pragma unroll
    for (int c = 0; c < 64; ++c) qacc = fmaf(pv[c], Wmlp[c * n_out + o], qacc);
    qout[(size_t)pt * n_out + o] = qacc;
    atomicAdd(&lsum[o], qacc);
    atomicAdd(&lsq[o], qacc * qacc);
  }
  __syncthreads();
  if (tid < n_out) { atomicAdd(&gsum[tid], lsum[tid]); atomicAdd(&gsq[tid], lsq[tid]); }
}

// ---------------------------------------------------------------------------
// final: relu(bn(q2)) @ W_mlp2 + b + bn(short) -> leaky 0.01 -> (B,128,N,1)
// ---------------------------------------------------------------------------
__global__ void final_kernel(const float* __restrict__ q2,
                             const float* __restrict__ sc2, const float* __restrict__ sh2,
                             const float* __restrict__ W2, const float* __restrict__ b2,
                             const float* __restrict__ ss,
                             const float* __restrict__ scs, const float* __restrict__ shs,
                             float* __restrict__ out) {
  __shared__ float sW[64 * 128];
  __shared__ float sb[128];
  const int tid = threadIdx.x;
  for (int i = tid; i < 8192; i += 256) sW[i] = W2[i];
  if (tid < 128) sb[tid] = b2[tid];
  __syncthreads();
  const int pt = blockIdx.x * 256 + tid;
  const int b = pt / NPTS, n = pt % NPTS;
  float x[64];
#pragma unroll
  for (int c = 0; c < 64; ++c)
    x[c] = fmaxf(fmaf(q2[(size_t)pt * 64 + c], sc2[c], sh2[c]), 0.f);
  for (int o = 0; o < 128; ++o) {
    float acc = sb[o];
#pragma unroll
    for (int c = 0; c < 64; ++c) acc = fmaf(x[c], sW[c * 128 + o], acc);
    acc += fmaf(ss[(size_t)pt * 128 + o], scs[o], shs[o]);
    acc = (acc >= 0.f) ? acc : 0.01f * acc;
    out[((size_t)b * 128 + o) * NPTS + n] = acc;
  }
}

// ---------------------------------------------------------------------------
extern "C" void kernel_launch(void* const* d_in, const int* in_sizes, int n_in,
                              void* d_out, int out_size, void* d_ws, size_t ws_size,
                              hipStream_t stream) {
  (void)in_sizes; (void)n_in; (void)out_size; (void)ws_size;
  const float* coords     = (const float*)d_in[0];
  const float* features   = (const float*)d_in[1];
  const float* W_mlp1     = (const float*)d_in[2];
  const float* b_mlp1     = (const float*)d_in[3];
  const float* W_lse1     = (const float*)d_in[4];
  const float* b_lse1     = (const float*)d_in[5];
  const float* g_lse1     = (const float*)d_in[6];
  const float* be_lse1    = (const float*)d_in[7];
  const float* W_att1_lin = (const float*)d_in[8];
  const float* W_att1_mlp = (const float*)d_in[9];
  const float* b_att1_mlp = (const float*)d_in[10];
  const float* g_att1     = (const float*)d_in[11];
  const float* be_att1    = (const float*)d_in[12];
  const float* W_lse2     = (const float*)d_in[13];
  const float* b_lse2     = (const float*)d_in[14];
  const float* g_lse2     = (const float*)d_in[15];
  const float* be_lse2    = (const float*)d_in[16];
  const float* W_att2_lin = (const float*)d_in[17];
  const float* W_att2_mlp = (const float*)d_in[18];
  const float* b_att2_mlp = (const float*)d_in[19];
  const float* g_att2     = (const float*)d_in[20];
  const float* be_att2    = (const float*)d_in[21];
  const float* W_mlp2     = (const float*)d_in[22];
  const float* b_mlp2     = (const float*)d_in[23];
  const float* W_short    = (const float*)d_in[24];
  const float* b_short    = (const float*)d_in[25];
  const float* g_short    = (const float*)d_in[26];
  const float* be_short   = (const float*)d_in[27];

  float* ws = (float*)d_ws;
  int*   idx    = (int*)d_ws;           // IDX_OFF == 0
  float* dist   = ws + DIST_OFF;
  float* xh     = ws + XH_OFF;
  float* ybuf   = ws + Y_OFF;
  float* q1     = ws + Q1_OFF;
  float* q2     = ws + Q2_OFF;
  float* sshort = ws + SS_OFF;
  float* S  = ws + STATS_OFF;   // raw sums
  float* SC = ws + SCSH_OFF;    // scale/shift
  float *sum_lse1 = S + 0,   *sq_lse1 = S + 32;
  float *sum_att1 = S + 64,  *sq_att1 = S + 96;
  float *sum_lse2 = S + 128, *sq_lse2 = S + 160;
  float *sum_att2 = S + 192, *sq_att2 = S + 256;
  float *sum_sh   = S + 320, *sq_sh   = S + 448;
  float *sc_lse1 = SC + 0,   *sh_lse1 = SC + 32;
  float *sc_att1 = SC + 64,  *sh_att1 = SC + 96;
  float *sc_lse2 = SC + 128, *sh_lse2 = SC + 160;
  float *sc_att2 = SC + 192, *sh_att2 = SC + 256;
  float *sc_sh   = SC + 320, *sh_sh   = SC + 448;
  _Float16* wh1 = (_Float16*)(ws + WH_OFF);
  _Float16* wh2 = wh1 + 4096;

  const float inv_bnk = 1.f / (float)(BATCH * NPTS * KNBR);  // 262144
  const float inv_bn  = 1.f / (float)(BATCH * NPTS);         // 16384

  prep_kernel<<<32, 256, 0, stream>>>(W_att1_lin, W_att2_lin, wh1, wh2, S);
  knn_kernel<<<BATCH * (NPTS / 256), 256, 0, stream>>>(coords, idx, dist);
  point_mlp_kernel<<<BATCH * NPTS / 256, 256, 0, stream>>>(
      features, W_mlp1, b_mlp1, W_short, b_short, xh, sshort, sum_sh, sq_sh);
  bn_fin_kernel<<<1, 128, 0, stream>>>(sum_sh, sq_sh, g_short, be_short, sc_sh, sh_sh, inv_bn, 128);

  // ---- stage 1 ----
  locse_kernel<<<BATCH * NPTS * KNBR / 256, 256, 0, stream>>>(
      coords, idx, dist, W_lse1, b_lse1, ybuf, sum_lse1, sq_lse1);
  bn_fin_kernel<<<1, 128, 0, stream>>>(sum_lse1, sq_lse1, g_lse1, be_lse1, sc_lse1, sh_lse1, inv_bnk, 32);
  att_pool_kernel<<<BATCH * NPTS / 8, 256, 0, stream>>>(
      ybuf, sc_lse1, sh_lse1, xh, nullptr, nullptr, wh1,
      W_att1_mlp, b_att1_mlp, q1, sum_att1, sq_att1, 32);
  bn_fin_kernel<<<1, 128, 0, stream>>>(sum_att1, sq_att1, g_att1, be_att1, sc_att1, sh_att1, inv_bn, 32);

  // ---- stage 2 (reuses ybuf) ----
  locse_kernel<<<BATCH * NPTS * KNBR / 256, 256, 0, stream>>>(
      coords, idx, dist, W_lse2, b_lse2, ybuf, sum_lse2, sq_lse2);
  bn_fin_kernel<<<1, 128, 0, stream>>>(sum_lse2, sq_lse2, g_lse2, be_lse2, sc_lse2, sh_lse2, inv_bnk, 32);
  att_pool_kernel<<<BATCH * NPTS / 8, 256, 0, stream>>>(
      ybuf, sc_lse2, sh_lse2, q1, sc_att1, sh_att1, wh2,
      W_att2_mlp, b_att2_mlp, q2, sum_att2, sq_att2, 64);
  bn_fin_kernel<<<1, 128, 0, stream>>>(sum_att2, sq_att2, g_att2, be_att2, sc_att2, sh_att2, inv_bn, 64);

  // ---- output ----
  final_kernel<<<BATCH * NPTS / 256, 256, 0, stream>>>(
      q2, sc_att2, sh_att2, W_mlp2, b_mlp2, sshort, sc_sh, sh_sh, (float*)d_out);
}
